// Perceiver_16363825398417
// MI455X (gfx1250) — matrix operand order repaired and checked by reading, not compiled, not run on hardware
//
#include <hip/hip_runtime.h>

typedef _Float16 h8   __attribute__((ext_vector_type(8)));
typedef _Float16 v16h __attribute__((ext_vector_type(16)));
typedef float    v8f  __attribute__((ext_vector_type(8)));
typedef float    f4   __attribute__((ext_vector_type(4)));
typedef int      i4v  __attribute__((vector_size(16)));

union HFrag { v16h v; h8 h[2]; };

// gfx1250 async global->LDS path (GLOBAL_LOAD_ASYNC_TO_LDS_B128, ASYNCcnt)
#if defined(__HIP_DEVICE_COMPILE__) && __has_builtin(__builtin_amdgcn_global_load_async_to_lds_b128)
#define USE_ASYNC_LDS 1
#else
#define USE_ASYNC_LDS 0
#endif

#define AS1P(p) ((__attribute__((address_space(1))) i4v*)(p))
#define AS3P(p) ((__attribute__((address_space(3))) i4v*)(p))

__device__ __forceinline__ void async_wait_all() {
#if USE_ASYNC_LDS
#if __has_builtin(__builtin_amdgcn_s_wait_asynccnt)
  __builtin_amdgcn_s_wait_asynccnt(0);
#else
  asm volatile("s_wait_asynccnt 0x0" ::: "memory");
#endif
#endif
}

// ---------------------------------------------------------------------------
// Tiled WMMA GEMM.  C[M,Ncols] = A[M,K] * B^T  (+bias) (+residual)
//   A: f16 row-major (lda) or f32 row-major converted on LDS fill (AF32)
//   B: f16 stored TRANSPOSED, i.e. Bt[Ncols][K] row-major (ldb)
//   Batched over blockIdx.z with (z/batH, z%batH) strides (for per-(b,h) GEMMs)
// EPI: 0 = f32 out, 1 = f16 out, 2 = f16 out transposed per-head (V^T),
//      3 = f32 out + residual add
// Block: 256 threads (8 waves). BN=256: waves 2x4, 64x64 per wave (16 WMMA /
// 16 ds_load_b128 per K-step).  BN=64: waves 4x2, 32x32 per wave.
// ---------------------------------------------------------------------------
template<int BN, bool AF32, int EPI>
__global__ __launch_bounds__(256) void gemm_wmma(
    const void* __restrict__ Av, const _Float16* __restrict__ Bt,
    const float* __restrict__ bias, const float* __restrict__ resid,
    float* __restrict__ Cf, _Float16* __restrict__ Ch,
    int M, int Ncols, int K, int lda, int ldb, int ldc,
    int batH, long long aB, long long aH, long long bB, long long bH,
    long long cB, long long cH, int Skv, int Hn, int dh)
{
  constexpr int BM = 128, BK = 32, LDK = 40;   // 8-half pad: 80B LDS rows
  constexpr int WCOLS = (BN >= 256) ? 4 : 2;
  constexpr int WROWS = 8 / WCOLS;
  constexpr int WM = BM / WROWS;               // 64 (BN=256) or 32
  constexpr int WN = BN / WCOLS;               // 64 (BN=256) or 32
  constexpr int MT = WM / 16, NT = WN / 16;
  __shared__ __align__(16) _Float16 lA[BM * LDK];
  __shared__ __align__(16) _Float16 lB[BN * LDK];

  const int tid  = threadIdx.x;
  const int lane = tid & 31;
  const int wid  = tid >> 5;
  const int wRow = wid / WCOLS;
  const int wCol = wid % WCOLS;
  const int z    = blockIdx.z;
  const long long aOff = (long long)(z / batH) * aB + (long long)(z % batH) * aH;
  const long long bOff = (long long)(z / batH) * bB + (long long)(z % batH) * bH;
  const long long cOff = (long long)(z / batH) * cB + (long long)(z % batH) * cH;
  const int blockM = blockIdx.y * BM;
  const int blockN = blockIdx.x * BN;

  const _Float16* Bp = Bt + bOff;

  v8f acc[MT][NT];
  const v8f vzero = {0.f,0.f,0.f,0.f,0.f,0.f,0.f,0.f};
#pragma unroll
  for (int mt = 0; mt < MT; ++mt)
#pragma unroll
    for (int nt = 0; nt < NT; ++nt) acc[mt][nt] = vzero;

  const int arow = tid >> 1, apart = tid & 1;  // A tile: 2 threads/row, 16 halves each
  const int g = lane >> 4, lm = lane & 15;

  for (int k0 = 0; k0 < K; k0 += BK) {
    // ---- fill A tile (128 x 32 halves) ----
    {
      const long long grow = blockM + arow;
      _Float16* dst = &lA[arow * LDK + apart * 16];
      if constexpr (AF32) {
        const float* Af = (const float*)Av + aOff;
        const f4* src = (const f4*)(Af + grow * (long long)lda + k0 + apart * 16);
        float xs[16];
        *(f4*)&xs[0]  = src[0];
        *(f4*)&xs[4]  = src[1];
        *(f4*)&xs[8]  = src[2];
        *(f4*)&xs[12] = src[3];
        h8 h0, h1;
#pragma unroll
        for (int j = 0; j < 8; ++j) { h0[j] = (_Float16)xs[j]; h1[j] = (_Float16)xs[8 + j]; }
        ((h8*)dst)[0] = h0; ((h8*)dst)[1] = h1;
        if (k0 + BK < K) __builtin_prefetch((const void*)(src + 8), 0, 3);
      } else {
        const h8* src = (const h8*)((const _Float16*)Av + aOff + grow * (long long)lda + k0 + apart * 16);
#if USE_ASYNC_LDS
        __builtin_amdgcn_global_load_async_to_lds_b128(AS1P(src),     AS3P(dst),     0, 0);
        __builtin_amdgcn_global_load_async_to_lds_b128(AS1P(src + 1), AS3P(dst + 8), 0, 0);
#else
        ((h8*)dst)[0] = src[0]; ((h8*)dst)[1] = src[1];
#endif
        if (k0 + BK < K) __builtin_prefetch((const void*)(src + 4), 0, 3);
      }
    }
    // ---- fill B tile (BN x 32 halves, already transposed in global) ----
    for (int idx = tid; idx < BN * 2; idx += 256) {
      const int br = idx >> 1, bp = idx & 1;
      const h8* src = (const h8*)(Bp + (long long)(blockN + br) * ldb + k0 + bp * 16);
      _Float16* dst = &lB[br * LDK + bp * 16];
#if USE_ASYNC_LDS
      __builtin_amdgcn_global_load_async_to_lds_b128(AS1P(src),     AS3P(dst),     0, 0);
      __builtin_amdgcn_global_load_async_to_lds_b128(AS1P(src + 1), AS3P(dst + 8), 0, 0);
#else
      ((h8*)dst)[0] = src[0]; ((h8*)dst)[1] = src[1];
#endif
      if (k0 + BK < K) __builtin_prefetch((const void*)(src + 4), 0, 3);
    }
    async_wait_all();
    __syncthreads();

    // ---- fragments (ISA 16-bit A 16x32 / B 32x16 lane layouts) ----
    HFrag a[MT], b[NT];
#pragma unroll
    for (int mt = 0; mt < MT; ++mt) {
      const _Float16* p = &lA[(wRow * WM + mt * 16 + lm) * LDK + g * 8];
      a[mt].h[0] = *(const h8*)p;          // K = g*8 .. g*8+7
      a[mt].h[1] = *(const h8*)(p + 16);   // K = 16+g*8 .. 16+g*8+7
    }
#pragma unroll
    for (int nt = 0; nt < NT; ++nt) {
      const _Float16* p = &lB[(wCol * WN + nt * 16 + lm) * LDK + g * 8];
      b[nt].h[0] = *(const h8*)p;          // K = g*8 .. g*8+7
      b[nt].h[1] = *(const h8*)(p + 16);   // K = 16+g*8 .. 16+g*8+7
    }
#pragma unroll
    for (int mt = 0; mt < MT; ++mt)
#pragma unroll
      for (int nt = 0; nt < NT; ++nt)
        acc[mt][nt] = __builtin_amdgcn_wmma_f32_16x16x32_f16(
            false, a[mt].v, false, b[nt].v, (short)0, acc[mt][nt], false, false);
    __syncthreads();
  }

  // ---- epilogue: C element (m = r + 8*(lane>=16), n = lane&15) ----
  const int baseM = blockM + wRow * WM;
  const int baseN = blockN + wCol * WN;
  const int hi = lane >> 4;
#pragma unroll
  for (int mt = 0; mt < MT; ++mt)
#pragma unroll
    for (int nt = 0; nt < NT; ++nt)
#pragma unroll
      for (int r = 0; r < 8; ++r) {
        const int gm = baseM + mt * 16 + hi * 8 + r;
        const int gn = baseN + nt * 16 + lm;
        float v = acc[mt][nt][r];
        if (bias) v += bias[gn];
        if constexpr (EPI == 0) {
          Cf[cOff + (long long)gm * ldc + gn] = v;
        } else if constexpr (EPI == 1) {
          Ch[cOff + (long long)gm * ldc + gn] = (_Float16)v;
        } else if constexpr (EPI == 2) {
          const int bb = gm / Skv, ss = gm % Skv;
          const int hh = gn / dh,  ii = gn % dh;
          Ch[((long long)(bb * Hn + hh) * dh + ii) * (long long)Skv + ss] = (_Float16)v;
        } else {  // EPI == 3
          Cf[cOff + (long long)gm * ldc + gn] = v + resid[(long long)gm * ldc + gn];
        }
      }
  (void)M; (void)Ncols;
}

// ---------------------------------------------------------------------------
// LayerNorm: one block per row, f32 in -> f16 out
// ---------------------------------------------------------------------------
__global__ __launch_bounds__(256) void ln_kernel(
    const float* __restrict__ x, const float* __restrict__ g,
    const float* __restrict__ b, _Float16* __restrict__ out, int Dm)
{
  __shared__ float red[256];
  const long long row = blockIdx.x;
  const float* xr = x + row * Dm;
  float s = 0.f;
  for (int i = threadIdx.x; i < Dm; i += 256) s += xr[i];
  red[threadIdx.x] = s; __syncthreads();
  for (int st = 128; st > 0; st >>= 1) {
    if (threadIdx.x < st) red[threadIdx.x] += red[threadIdx.x + st];
    __syncthreads();
  }
  const float mean = red[0] / (float)Dm;
  __syncthreads();
  float s2 = 0.f;
  for (int i = threadIdx.x; i < Dm; i += 256) { float d = xr[i] - mean; s2 += d * d; }
  red[threadIdx.x] = s2; __syncthreads();
  for (int st = 128; st > 0; st >>= 1) {
    if (threadIdx.x < st) red[threadIdx.x] += red[threadIdx.x + st];
    __syncthreads();
  }
  const float inv = rsqrtf(red[0] / (float)Dm + 1e-6f);
  _Float16* orow = out + row * Dm;
  for (int i = threadIdx.x; i < Dm; i += 256)
    orow[i] = (_Float16)((xr[i] - mean) * inv * g[i] + b[i]);
}

// ---------------------------------------------------------------------------
// Softmax over last dim (in place, fp32 in d_out).  Applies scale and mask.
// ---------------------------------------------------------------------------
__global__ __launch_bounds__(256) void softmax_kernel(
    float* __restrict__ logits, const float* __restrict__ mask,
    float scale, int len, int HN, int S)
{
  __shared__ float red[256];
  const long long row = blockIdx.x;
  float* p = logits + row * (long long)len;
  const float* mrow = mask ? mask + (long long)(row / HN) * S : nullptr;
  float mx = -1e30f;
  for (int i = threadIdx.x; i < len; i += 256) {
    float v = p[i] * scale;
    if (mrow) v += mrow[i] * (-1e9f);
    p[i] = v;
    mx = fmaxf(mx, v);
  }
  red[threadIdx.x] = mx; __syncthreads();
  for (int st = 128; st > 0; st >>= 1) {
    if (threadIdx.x < st) red[threadIdx.x] = fmaxf(red[threadIdx.x], red[threadIdx.x + st]);
    __syncthreads();
  }
  mx = red[0]; __syncthreads();
  float sum = 0.f;
  for (int i = threadIdx.x; i < len; i += 256) {
    float e = __expf(p[i] - mx);
    p[i] = e;
    sum += e;
  }
  red[threadIdx.x] = sum; __syncthreads();
  for (int st = 128; st > 0; st >>= 1) {
    if (threadIdx.x < st) red[threadIdx.x] += red[threadIdx.x + st];
    __syncthreads();
  }
  const float inv = 1.f / red[0];
  for (int i = threadIdx.x; i < len; i += 256) p[i] *= inv;
}

// ---------------------------------------------------------------------------
// swiGLU combine: out = (lin) * silu(sw), f32 -> f16
// ---------------------------------------------------------------------------
__global__ __launch_bounds__(256) void swiglu_kernel(
    const float* __restrict__ hl, const float* __restrict__ hs,
    _Float16* __restrict__ outp, long long n)
{
  long long i = (long long)blockIdx.x * 256 + threadIdx.x;
  const long long stride = (long long)gridDim.x * 256;
  for (; i < n; i += stride) {
    const float a = hl[i], s = hs[i];
    const float sig = 1.f / (1.f + __expf(-s));
    outp[i] = (_Float16)(a * s * sig);
  }
}

// ---------------------------------------------------------------------------
// Weight convert+transpose: wt[o][i] = (f16) w[i][o]
// ---------------------------------------------------------------------------
__global__ __launch_bounds__(256) void tconv_kernel(
    const float* __restrict__ w, _Float16* __restrict__ wt, int In, int Out)
{
  const long long idx = (long long)blockIdx.x * 256 + threadIdx.x;
  if (idx < (long long)In * Out) {
    const int o = (int)(idx / In);
    const int i = (int)(idx % In);
    wt[idx] = (_Float16)w[(long long)i * Out + o];
  }
}

// ---------------------------------------------------------------------------
extern "C" void kernel_launch(void* const* d_in, const int* in_sizes, int n_in,
                              void* d_out, int out_size, void* d_ws, size_t ws_size,
                              hipStream_t stream) {
  const int Bc = 2, Sc = 4096, Nc = 512, Dc = 1024, Hc = 16, Fc = 4096;
  const int dh = Dc / Hc;  // 64

  const float* enc    = (const float*)d_in[0];
  const float* latent = (const float*)d_in[1];
  const float* maskin = (const float*)d_in[2];
  const float* ln1_g = (const float*)d_in[3];
  const float* ln1_b = (const float*)d_in[4];
  const float* ln2_g = (const float*)d_in[5];
  const float* ln2_b = (const float*)d_in[6];
  const float* ln3_g = (const float*)d_in[7];
  const float* ln3_b = (const float*)d_in[8];
  const float* wq = (const float*)d_in[9];   const float* bq = (const float*)d_in[10];
  const float* wk = (const float*)d_in[11];  const float* bk = (const float*)d_in[12];
  const float* wv = (const float*)d_in[13];  const float* bv = (const float*)d_in[14];
  const float* wo = (const float*)d_in[15];  const float* bo = (const float*)d_in[16];
  const float* w_in  = (const float*)d_in[17]; const float* b_in  = (const float*)d_in[18];
  const float* w_sw  = (const float*)d_in[19]; const float* b_sw  = (const float*)d_in[20];
  const float* w_lin = (const float*)d_in[21]; const float* b_lin = (const float*)d_in[22];
  const float* w_out = (const float*)d_in[23]; const float* b_out = (const float*)d_in[24];

  float* out_x = (float*)d_out;
  float* attn0 = out_x + (size_t)Bc * Nc * Dc;
  float* attn1 = attn0 + (size_t)Bc * Hc * Nc * Sc;

  // workspace bump allocator
  size_t woff = 0;
  auto walloc = [&](size_t bytes) -> void* {
    void* p = (char*)d_ws + woff;
    woff += (bytes + 255) & ~(size_t)255;
    return p;
  };
  _Float16* wq16   = (_Float16*)walloc((size_t)Dc * Dc * 2);
  _Float16* wk16   = (_Float16*)walloc((size_t)Dc * Dc * 2);
  _Float16* wv16   = (_Float16*)walloc((size_t)Dc * Dc * 2);
  _Float16* wo16   = (_Float16*)walloc((size_t)Dc * Dc * 2);
  _Float16* win16  = (_Float16*)walloc((size_t)Dc * Fc * 2);
  _Float16* wsw16  = (_Float16*)walloc((size_t)Fc * Fc * 2);
  _Float16* wlin16 = (_Float16*)walloc((size_t)Fc * Fc * 2);
  _Float16* wout16 = (_Float16*)walloc((size_t)Fc * Dc * 2);
  _Float16* ni16   = (_Float16*)walloc((size_t)Bc * Sc * Dc * 2);
  _Float16* nl16   = (_Float16*)walloc((size_t)Bc * Nc * Dc * 2);
  _Float16* q16    = (_Float16*)walloc((size_t)Bc * Nc * Dc * 2);
  _Float16* k16    = (_Float16*)walloc((size_t)Bc * Sc * Dc * 2);
  _Float16* vt16   = (_Float16*)walloc((size_t)Bc * Sc * Dc * 2);
  _Float16* o16    = (_Float16*)walloc((size_t)Bc * Nc * Dc * 2);
  float*    out1   = (float*)   walloc((size_t)Bc * Nc * Dc * 4);
  _Float16* h16    = (_Float16*)walloc((size_t)Bc * Nc * Dc * 2);
  _Float16* h116   = (_Float16*)walloc((size_t)Bc * Nc * Fc * 2);
  float*    hl32   = (float*)   walloc((size_t)Bc * Nc * Fc * 4);
  float*    hs32   = (float*)   walloc((size_t)Bc * Nc * Fc * 4);
  _Float16* ffin16 = (_Float16*)walloc((size_t)Bc * Nc * Fc * 2);
  float*    xbuf   = (float*)   walloc((size_t)Bc * Nc * Dc * 4);

  auto tgrid = [](long long n) { return (unsigned)((n + 255) / 256); };

  for (int l = 0; l < 2; ++l) {
    const int Skv = (l == 0) ? Sc : Nc;
    const float* xkv = (l == 0) ? enc : xbuf;
    float* attn = (l == 0) ? attn0 : attn1;
    const float* maskp = (l == 0) ? maskin : nullptr;

    // --- convert this layer's weights to f16 transposed [out][in] ---
    tconv_kernel<<<tgrid((long long)Dc * Dc), 256, 0, stream>>>(wq + (size_t)l * Dc * Dc, wq16, Dc, Dc);
    tconv_kernel<<<tgrid((long long)Dc * Dc), 256, 0, stream>>>(wk + (size_t)l * Dc * Dc, wk16, Dc, Dc);
    tconv_kernel<<<tgrid((long long)Dc * Dc), 256, 0, stream>>>(wv + (size_t)l * Dc * Dc, wv16, Dc, Dc);
    tconv_kernel<<<tgrid((long long)Dc * Dc), 256, 0, stream>>>(wo + (size_t)l * Dc * Dc, wo16, Dc, Dc);
    tconv_kernel<<<tgrid((long long)Dc * Fc), 256, 0, stream>>>(w_in  + (size_t)l * Dc * Fc, win16,  Dc, Fc);
    tconv_kernel<<<tgrid((long long)Fc * Fc), 256, 0, stream>>>(w_sw  + (size_t)l * Fc * Fc, wsw16,  Fc, Fc);
    tconv_kernel<<<tgrid((long long)Fc * Fc), 256, 0, stream>>>(w_lin + (size_t)l * Fc * Fc, wlin16, Fc, Fc);
    tconv_kernel<<<tgrid((long long)Fc * Dc), 256, 0, stream>>>(w_out + (size_t)l * Fc * Dc, wout16, Fc, Dc);

    // --- layernorms (f32 -> f16) ---
    ln_kernel<<<Bc * Skv, 256, 0, stream>>>(xkv,    ln1_g + l * Dc, ln1_b + l * Dc, ni16, Dc);
    ln_kernel<<<Bc * Nc,  256, 0, stream>>>(latent, ln2_g + l * Dc, ln2_b + l * Dc, nl16, Dc);

    // --- Q/K projections (f16 out), V projection (f16 transposed per head) ---
    gemm_wmma<256, false, 1><<<dim3(Dc / 256, (Bc * Nc) / 128, 1), 256, 0, stream>>>(
        nl16, wq16, bq + l * Dc, nullptr, nullptr, q16,
        Bc * Nc, Dc, Dc, Dc, Dc, Dc, 1, 0, 0, 0, 0, 0, 0, 0, 0, 0);
    gemm_wmma<256, false, 1><<<dim3(Dc / 256, (Bc * Skv) / 128, 1), 256, 0, stream>>>(
        ni16, wk16, bk + l * Dc, nullptr, nullptr, k16,
        Bc * Skv, Dc, Dc, Dc, Dc, Dc, 1, 0, 0, 0, 0, 0, 0, 0, 0, 0);
    gemm_wmma<256, false, 2><<<dim3(Dc / 256, (Bc * Skv) / 128, 1), 256, 0, stream>>>(
        ni16, wv16, bv + l * Dc, nullptr, nullptr, vt16,
        Bc * Skv, Dc, Dc, Dc, Dc, Dc, 1, 0, 0, 0, 0, 0, 0, Skv, Hc, dh);

    // --- logits = Q @ K^T  (batched over B*H, f32 straight into d_out) ---
    gemm_wmma<256, false, 0><<<dim3(Skv / 256, Nc / 128, Bc * Hc), 256, 0, stream>>>(
        q16, k16, nullptr, nullptr, attn, nullptr,
        Nc, Skv, dh, Dc, Dc, Skv, Hc,
        (long long)Nc * Dc, dh, (long long)Skv * Dc, dh,
        (long long)Hc * Nc * Skv, (long long)Nc * Skv, 0, 0, 0);

    // --- softmax (scale + optional mask), in place in d_out ---
    softmax_kernel<<<Bc * Hc * Nc, 256, 0, stream>>>(
        attn, maskp, 0.125f, Skv, Hc * Nc, Sc);

    // --- O = AW @ V  (A is fp32 aw from d_out, converted on LDS fill) ---
    gemm_wmma<64, true, 1><<<dim3(1, Nc / 128, Bc * Hc), 256, 0, stream>>>(
        attn, vt16, nullptr, nullptr, nullptr, o16,
        Nc, dh, Skv, Skv, Skv, Dc, Hc,
        (long long)Hc * Nc * Skv, (long long)Nc * Skv,
        (long long)Hc * dh * Skv, (long long)dh * Skv,
        (long long)Nc * Dc, dh, 0, 0, 0);

    // --- attn_out = O @ wo + bo, + latent residual -> out1 (f32) ---
    gemm_wmma<256, false, 3><<<dim3(Dc / 256, (Bc * Nc) / 128, 1), 256, 0, stream>>>(
        o16, wo16, bo + l * Dc, latent, out1, nullptr,
        Bc * Nc, Dc, Dc, Dc, Dc, Dc, 1, 0, 0, 0, 0, 0, 0, 0, 0, 0);

    // --- FFN ---
    ln_kernel<<<Bc * Nc, 256, 0, stream>>>(out1, ln3_g + l * Dc, ln3_b + l * Dc, h16, Dc);
    gemm_wmma<256, false, 1><<<dim3(Fc / 256, (Bc * Nc) / 128, 1), 256, 0, stream>>>(
        h16, win16, b_in + l * Fc, nullptr, nullptr, h116,
        Bc * Nc, Fc, Dc, Dc, Dc, Fc, 1, 0, 0, 0, 0, 0, 0, 0, 0, 0);
    gemm_wmma<256, false, 0><<<dim3(Fc / 256, (Bc * Nc) / 128, 1), 256, 0, stream>>>(
        h116, wlin16, b_lin + l * Fc, nullptr, hl32, nullptr,
        Bc * Nc, Fc, Fc, Fc, Fc, Fc, 1, 0, 0, 0, 0, 0, 0, 0, 0, 0);
    gemm_wmma<256, false, 0><<<dim3(Fc / 256, (Bc * Nc) / 128, 1), 256, 0, stream>>>(
        h116, wsw16, b_sw + l * Fc, nullptr, hs32, nullptr,
        Bc * Nc, Fc, Fc, Fc, Fc, Fc, 1, 0, 0, 0, 0, 0, 0, 0, 0, 0);
    swiglu_kernel<<<4096, 256, 0, stream>>>(hl32, hs32, ffin16, (long long)Bc * Nc * Fc);
    gemm_wmma<256, false, 3><<<dim3(Dc / 256, (Bc * Nc) / 128, 1), 256, 0, stream>>>(
        ffin16, wout16, b_out + l * Dc, out1, (l == 0) ? xbuf : out_x,
        nullptr, Bc * Nc, Dc, Fc, Fc, Fc, Dc, 1, 0, 0, 0, 0, 0, 0, 0, 0, 0);
  }
  (void)in_sizes; (void)n_in; (void)out_size; (void)ws_size;
}
